// LatentField_28346784153620
// MI455X (gfx1250) — matrix operand order repaired and checked by reading, not compiled, not run on hardware
//
#include <hip/hip_runtime.h>
#include <hip/hip_bf16.h>

typedef __attribute__((ext_vector_type(16))) _Float16 v16h;
typedef __attribute__((ext_vector_type(8)))  _Float16 v8h;
typedef __attribute__((ext_vector_type(8)))  float    v8f;

#define FEAT 64
#define HID  128
#define OUTD 4
#define BB   4
#define NN   200000
#define HH   512
#define WW   512
#define ASTRIDE 104    // aTile row stride in halves (96 padded K + 8)
#define FSTRIDE 136    // fTile row stride in halves (128 + 8)
#define WAVES 4
#define NTILES (BB*NN/16)   // 50000, exact (800000 % 16 == 0)

// workspace layout (bytes)
#define TWS_BYTES ((size_t)BB*HH*WW*FEAT*2)       // 128MB transposed f16 grid (fits in 192MB L2)
#define W1F_OFF   TWS_BYTES
#define W1F_HALVES (24*32*16)                     // 24 frags x 32 lanes x 16 halves
#define W2F_OFF   (W1F_OFF + (size_t)W1F_HALVES*2)
#define W2F_HALVES (4*32*16)

__device__ __forceinline__ float readlane_f(float v, int l) {
  return __int_as_float(__builtin_amdgcn_readlane(__float_as_int(v), l));
}

// ---------------- grid transpose: (B,C,H,W) f32 -> (B,H,W,C) f16 ----------------
__global__ __launch_bounds__(256) void k_transpose(const float* __restrict__ grid,
                                                   _Float16* __restrict__ tws) {
  __shared__ _Float16 tile[64*68];   // [x][c], row stride 68 halves (136B, dword-aligned)
  int by = blockIdx.x;               // 0..B*H-1
  int b = by / HH, y = by % HH;
  const float* gb = grid + (size_t)b*FEAT*HH*WW + (size_t)y*WW;
  _Float16* ob = tws + (((size_t)b*HH + y)*WW)*FEAT;
  int t = threadIdx.x;
  for (int xb = 0; xb < WW; xb += 64) {
    #pragma unroll
    for (int i = 0; i < 16; ++i) {                 // read 64x * 64c, coalesced in x
      int e = i*256 + t;
      int c = e >> 6, x = e & 63;
      tile[x*68 + c] = (_Float16)gb[(size_t)c*HH*WW + xb + x];
    }
    __syncthreads();
    #pragma unroll
    for (int i = 0; i < 8; ++i) {                  // write packed f16x2, contiguous in c
      int e = i*256 + t;
      int c2 = e & 31, x = e >> 5;
      unsigned int v = *(const unsigned int*)&tile[x*68 + 2*c2];
      *(unsigned int*)&ob[(size_t)(xb + x)*FEAT + 2*c2] = v;
    }
    __syncthreads();
  }
}

// ------- pre-swizzle weights into WMMA B-fragment per-lane layout (f16) -------
// 16-bit B (32x16): half h of lane half hh -> K = 8*hh + (h&7) + (h>=8 ? 16 : 0).
__global__ void k_frags(const float* __restrict__ w1, const float* __restrict__ w2,
                        _Float16* __restrict__ w1f, _Float16* __restrict__ w2f) {
  int t = blockIdx.x*blockDim.x + threadIdx.x;
  int stride = gridDim.x*blockDim.x;
  for (int e = t; e < 24*32*16; e += stride) {     // frag f = k*8+n
    int h = e & 15, lane = (e >> 4) & 31, f = e >> 9;
    int k = f >> 3, n = f & 7;
    int K = k*32 + ((lane < 16) ? 0 : 8) + (h & 7) + ((h & 8) ? 16 : 0);
    int Nc = n*16 + (lane & 15);
    float v = (K < FEAT+1) ? w1[K*HID + Nc] : 0.f;
    w1f[e] = (_Float16)v;
  }
  for (int e = t; e < 4*32*16; e += stride) {      // frag index = k
    int h = e & 15, lane = (e >> 4) & 31, k = e >> 9;
    int K = k*32 + ((lane < 16) ? 0 : 8) + (h & 7) + ((h & 8) ? 16 : 0);
    int Nc = lane & 15;
    float v = (Nc < OUTD) ? w2[K*OUTD + Nc] : 0.f;
    w2f[e] = (_Float16)v;
  }
}

// ---------------- main: gather + bilinear + WMMA MLP ----------------
__global__ __launch_bounds__(128) void k_main(
    const float* __restrict__ xyz, const _Float16* __restrict__ tws,
    const _Float16* __restrict__ w1f, const _Float16* __restrict__ w2f,
    const float* __restrict__ b1, const float* __restrict__ b2,
    float* __restrict__ out)
{
  __shared__ alignas(32) _Float16 sW1[24*32*16];   // 24KB
  __shared__ alignas(32) _Float16 sW2[4*32*16];    // 4KB
  __shared__ float sB1[HID];
  __shared__ float sB2[OUTD];
  __shared__ alignas(16) _Float16 sA[WAVES][16*ASTRIDE];
  __shared__ alignas(16) _Float16 sF[WAVES][16*FSTRIDE];
  __shared__ alignas(16) float    sO[WAVES][16*OUTD];

  int t = threadIdx.x;
  // block-level preload of fragment tables + biases
  {
    const uint4* s1 = (const uint4*)w1f; uint4* d1 = (uint4*)sW1;
    for (int i = t; i < 24*32*16*2/16; i += blockDim.x) d1[i] = s1[i];
    const uint4* s2 = (const uint4*)w2f; uint4* d2 = (uint4*)sW2;
    for (int i = t; i < 4*32*16*2/16; i += blockDim.x) d2[i] = s2[i];
    for (int i = t; i < HID; i += blockDim.x) sB1[i] = b1[i];
    if (t < OUTD) sB2[t] = b2[t];
  }
  __syncthreads();

  int wave = t >> 5, lane = t & 31;
  _Float16* aT = sA[wave];
  _Float16* fT = sF[wave];
  float*    oT = sO[wave];

  // zero the K-padding columns 65..95 once (never overwritten afterwards)
  for (int e = lane; e < 16*31; e += 32) {
    int r = e / 31, c = 65 + e % 31;
    aT[r*ASTRIDE + c] = (_Float16)0.f;
  }

  const unsigned int* twsU = (const unsigned int*)tws;
  int waveId  = blockIdx.x*WAVES + wave;
  int waveCnt = gridDim.x*WAVES;
  int p16 = lane & 15;
  int srow = (lane >= 16) ? 8 : 0;

  for (int tile = waveId; tile < NTILES; tile += waveCnt) {
    // ---- per-point setup (all lanes replicate lanes 0..15 -> no divergence) ----
    int g = tile*16 + p16;
    int b = g / NN;
    float px = xyz[3*g+0], pz = xyz[3*g+1], py = xyz[3*g+2];
    float fx = (px + 1.f)*0.5f*(float)(WW-1);
    float fy = (py + 1.f)*0.5f*(float)(HH-1);
    float x0f = floorf(fx), y0f = floorf(fy);
    float wx = fx - x0f,    wy = fy - y0f;
    int x0 = min(max((int)x0f, 0), WW-1);
    int x1 = min(x0+1, WW-1);
    int y0 = min(max((int)y0f, 0), HH-1);
    int y1 = min(y0+1, HH-1);
    int rowb = b*HH;
    int o00 = (((rowb+y0)*WW + x0)*FEAT) >> 1;     // dword (f16x2) units
    int o01 = (((rowb+y0)*WW + x1)*FEAT) >> 1;
    int o10 = (((rowb+y1)*WW + x0)*FEAT) >> 1;
    int o11 = (((rowb+y1)*WW + x1)*FEAT) >> 1;

    // ---- gather: per point, 4 corners x 128B, one wave-wide coalesced load each ----
    // readlane broadcast -> wave-uniform base (SGPR) + lane*4 offset: saddr global loads,
    // no ds_bpermute traffic competing with WMMA fragment loads.
    #pragma unroll
    for (int p = 0; p < 16; ++p) {
      const unsigned int* p00 = twsU + __builtin_amdgcn_readlane(o00, p);
      const unsigned int* p01 = twsU + __builtin_amdgcn_readlane(o01, p);
      const unsigned int* p10 = twsU + __builtin_amdgcn_readlane(o10, p);
      const unsigned int* p11 = twsU + __builtin_amdgcn_readlane(o11, p);
      float wxp = readlane_f(wx, p);
      float wyp = readlane_f(wy, p);
      unsigned int u00 = p00[lane];
      unsigned int u01 = p01[lane];
      unsigned int u10 = p10[lane];
      unsigned int u11 = p11[lane];
      union HU { unsigned int u; _Float16 h[2]; };
      HU c00, c01, c10, c11, rr;
      c00.u = u00; c01.u = u01; c10.u = u10; c11.u = u11;
      #pragma unroll
      for (int j = 0; j < 2; ++j) {
        float top = (float)c00.h[j]*(1.f-wxp) + (float)c01.h[j]*wxp;
        float bot = (float)c10.h[j]*(1.f-wxp) + (float)c11.h[j]*wxp;
        rr.h[j] = (_Float16)(top*(1.f-wyp) + bot*wyp);
      }
      *(unsigned int*)&aT[p*ASTRIDE + 2*lane] = rr.u;   // channels 2*lane..2*lane+1
    }
    // z feature, channel 64: lanes 16..31 duplicate lanes 0..15 (same addr, same data)
    aT[p16*ASTRIDE + 64] = (_Float16)pz;
    asm volatile("s_wait_dscnt 0" ::: "memory");        // wave-local LDS visibility

    // ---- layer 1: (16 x 96) @ (96 x 128), f16 WMMA, f32 accumulate ----
    v8f acc[8] = {};
    #pragma unroll
    for (int k = 0; k < 3; ++k) {
      // 16-bit A layout: lane holds M=lane%16; halves at K = k*32 + s..s+7 and +16..+23
      int abase = p16*ASTRIDE + k*32 + srow;
      v8h alo = *(const v8h*)&aT[abase];
      v8h ahi = *(const v8h*)&aT[abase + 16];
      v16h A = __builtin_shufflevector(alo, ahi, 0,1,2,3,4,5,6,7,8,9,10,11,12,13,14,15);
      // load all 8 B fragments first, then issue 8 back-to-back WMMAs
      v16h Bm[8];
      #pragma unroll
      for (int n = 0; n < 8; ++n)
        Bm[n] = *(const v16h*)&sW1[((k*8+n)*32 + lane)*16];
      #pragma unroll
      for (int n = 0; n < 8; ++n)
        acc[n] = __builtin_amdgcn_wmma_f32_16x16x32_f16(
            false, A, false, Bm[n], (short)0, acc[n], false, false);
    }

    // ---- bias + relu -> fT (f16), D layout: (M = r+srow, N = lane%16) ----
    #pragma unroll
    for (int n = 0; n < 8; ++n) {
      float bias = sB1[n*16 + p16];
      #pragma unroll
      for (int r2 = 0; r2 < 8; ++r2) {
        float v = acc[n][r2] + bias;
        v = v > 0.f ? v : 0.f;
        fT[(r2 + srow)*FSTRIDE + n*16 + p16] = (_Float16)v;
      }
    }
    asm volatile("s_wait_dscnt 0" ::: "memory");

    // ---- layer 2: (16 x 128) @ (128 x 16), 4 valid output columns ----
    v8f acc2 = {};
    {
      v16h A2[4], B2[4];
      #pragma unroll
      for (int k = 0; k < 4; ++k) {
        int abase = p16*FSTRIDE + k*32 + srow;
        v8h alo = *(const v8h*)&fT[abase];
        v8h ahi = *(const v8h*)&fT[abase + 16];
        A2[k] = __builtin_shufflevector(alo, ahi, 0,1,2,3,4,5,6,7,8,9,10,11,12,13,14,15);
        B2[k] = *(const v16h*)&sW2[(k*32 + lane)*16];
      }
      #pragma unroll
      for (int k = 0; k < 4; ++k)
        acc2 = __builtin_amdgcn_wmma_f32_16x16x32_f16(
            false, A2[k], false, B2[k], (short)0, acc2, false, false);
    }

    if (p16 < OUTD) {
      float bias = sB2[p16];
      #pragma unroll
      for (int r2 = 0; r2 < 8; ++r2)
        oT[(r2 + srow)*OUTD + p16] = acc2[r2] + bias;
    }
    asm volatile("s_wait_dscnt 0" ::: "memory");
    if (lane < 16) {                                // coalesced float4 stores
      float4 v = *(const float4*)&oT[lane*OUTD];
      *(float4*)&out[(size_t)(tile*16 + lane)*OUTD] = v;
    }
  }
}

extern "C" void kernel_launch(void* const* d_in, const int* in_sizes, int n_in,
                              void* d_out, int out_size, void* d_ws, size_t ws_size,
                              hipStream_t stream) {
  (void)in_sizes; (void)n_in; (void)out_size; (void)ws_size;
  const float* xyz  = (const float*)d_in[0];
  const float* grid = (const float*)d_in[1];
  const float* w1   = (const float*)d_in[2];
  const float* b1   = (const float*)d_in[3];
  const float* w2   = (const float*)d_in[4];
  const float* b2   = (const float*)d_in[5];
  float* out = (float*)d_out;

  char* ws = (char*)d_ws;
  _Float16* tws = (_Float16*)ws;                    // 128MB transposed grid
  _Float16* w1f = (_Float16*)(ws + W1F_OFF);        // 24KB layer-1 fragments
  _Float16* w2f = (_Float16*)(ws + W2F_OFF);        // 4KB layer-2 fragments

  k_transpose<<<BB*HH, 256, 0, stream>>>(grid, tws);
  k_frags<<<32, 256, 0, stream>>>(w1, w2, w1f, w2f);
  k_main<<<1024, 128, 0, stream>>>(xyz, tws, w1f, w2f, b1, b2, out);
}
